// LSTMnetwork_76982993813549
// MI455X (gfx1250) — matrix operand order repaired and checked by reading, not compiled, not run on hardware
//
#include <hip/hip_runtime.h>
#include <hip/hip_bf16.h>
#include <math.h>

#define INPUT   2048
#define HIDDEN  2048
#define GATES   8192   // 4*HIDDEN, PyTorch gate order i,f,g,o
#define T_STEPS 4096
#define DENSE_N 512

typedef __attribute__((ext_vector_type(16))) __bf16 v16bf;
typedef __attribute__((ext_vector_type(8)))  __bf16 v8bf;
typedef __attribute__((ext_vector_type(8)))  float  v8f;

// matches hipcc diagnostic: param type is 'int __attribute__((vector_size(16))) *'
typedef int v4i __attribute__((vector_size(4 * sizeof(int))));

static __device__ __forceinline__ unsigned short f2bf(float f) {
  unsigned u = __float_as_uint(f);
  unsigned r = u + 0x7fffu + ((u >> 16) & 1u);   // round-to-nearest-even
  return (unsigned short)(r >> 16);
}
static __device__ __forceinline__ float bflo(unsigned w) { return __uint_as_float(w << 16); }
static __device__ __forceinline__ float bfhi(unsigned w) { return __uint_as_float(w & 0xffff0000u); }
static __device__ __forceinline__ float sigmoidf_(float x) { return 1.0f / (1.0f + __expf(-x)); }

// ---------------- prep: fused bias, h0 seed, counter reset ----------------
__global__ void prep_kernel(const float* __restrict__ b_ih, const float* __restrict__ b_hh,
                            const float* __restrict__ h0,
                            float* __restrict__ bias, float* __restrict__ hbuf,
                            int* __restrict__ counter) {
  int i = blockIdx.x * blockDim.x + threadIdx.x;
  if (i < GATES)  bias[i] = b_ih[i] + b_hh[i];
  if (i < HIDDEN) hbuf[i] = h0[i];
  if (i == 0)     *counter = 0;
}

// ---------------- f32 -> bf16 conversion ----------------
__global__ void cvt_bf16_kernel(unsigned short* __restrict__ dst,
                                const float* __restrict__ src, int n) {
  int i = blockIdx.x * blockDim.x + threadIdx.x;
  if (i < n) dst[i] = f2bf(src[i]);
}

// ---------------- W_hh -> transposed, k-quad-packed bf16 ----------------
// Q[(k/4)*GATES + j] = { bf16 W_hh[j][k..k+3], k%4==0 } packed in uint2
__global__ void pack_whh_kernel(uint2* __restrict__ q, const float* __restrict__ whh) {
  int idx = blockIdx.x * blockDim.x + threadIdx.x;   // GATES * HIDDEN/4 threads
  int j  = idx & (GATES - 1);
  int kq = idx >> 13;
  const float4 w = *(const float4*)(whh + (size_t)j * HIDDEN + (size_t)kq * 4);
  uint2 o;
  o.x = (unsigned)f2bf(w.x) | ((unsigned)f2bf(w.y) << 16);
  o.y = (unsigned)f2bf(w.z) | ((unsigned)f2bf(w.w) << 16);
  q[(size_t)kq * GATES + j] = o;
}

// ---------------- WMMA GEMM: G = x1 @ W_ih^T + bias ----------------
// Async double-buffered tile staging via GLOBAL_LOAD_ASYNC_TO_LDS_B128 (ASYNCcnt).
#define BM 128
#define BN 128
#define BK 32
#define LDT 40   // padded LDS row stride in bf16 elements (80B, 16B aligned)

__global__ __launch_bounds__(256) void gemm_kernel(
    const unsigned short* __restrict__ Ab,   // x1 bf16, (T_STEPS x INPUT) row-major
    const unsigned short* __restrict__ Bb,   // W_ih bf16, (GATES x INPUT) row-major
    const float* __restrict__ bias,
    float* __restrict__ G) {                 // (T_STEPS x GATES) f32
  __shared__ __align__(16) unsigned short As[2][BM * LDT];
  __shared__ __align__(16) unsigned short Bs[2][BN * LDT];

  const int tid   = threadIdx.x;
  const int lane  = tid & 31;
  const int wid   = tid >> 5;    // 8 waves
  const int wm    = wid >> 2;    // 0..1  (64 rows each)
  const int wn    = wid & 3;     // 0..3  (32 cols each)
  const int bm0   = blockIdx.y * BM;
  const int bn0   = blockIdx.x * BN;
  const int lsub  = lane & 15;
  const int lhalf = lane >> 4;

  // staging geometry: 512 chunks of 16B per tile; thread handles chunks {tid, tid+256}
  const int srow0 = tid >> 2;
  const int srow1 = (tid + 256) >> 2;
  const int scol  = tid & 3;                 // (tid+256)&3 == tid&3

  // issue one K-tile's async loads (A and B) into LDS buffer `buf`
  auto stage = [&](int k0, int buf) {
    __builtin_amdgcn_global_load_async_to_lds_b128(
        (v4i*)(Ab + (size_t)(bm0 + srow0) * INPUT + k0 + scol * 8),
        (v4i*)&As[buf][srow0 * LDT + scol * 8], 0, 0);
    __builtin_amdgcn_global_load_async_to_lds_b128(
        (v4i*)(Ab + (size_t)(bm0 + srow1) * INPUT + k0 + scol * 8),
        (v4i*)&As[buf][srow1 * LDT + scol * 8], 0, 0);
    __builtin_amdgcn_global_load_async_to_lds_b128(
        (v4i*)(Bb + (size_t)(bn0 + srow0) * INPUT + k0 + scol * 8),
        (v4i*)&Bs[buf][srow0 * LDT + scol * 8], 0, 0);
    __builtin_amdgcn_global_load_async_to_lds_b128(
        (v4i*)(Bb + (size_t)(bn0 + srow1) * INPUT + k0 + scol * 8),
        (v4i*)&Bs[buf][srow1 * LDT + scol * 8], 0, 0);
  };

  v8f acc[4][2];
  #pragma unroll
  for (int mi = 0; mi < 4; ++mi)
    #pragma unroll
    for (int ni = 0; ni < 2; ++ni)
      #pragma unroll
      for (int e = 0; e < 8; ++e) acc[mi][ni][e] = 0.0f;

  stage(0, 0);

  const int NK = INPUT / BK;   // 64
  for (int i = 0; i < NK; ++i) {
    const int buf = i & 1;
    __builtin_amdgcn_s_wait_asynccnt(0);   // my async writes for `buf` have landed
    __syncthreads();                        // everyone's writes landed; buf^1 free
    if (i + 1 < NK) stage((i + 1) * BK, buf ^ 1);

    // A fragment: lane L<16 holds row M=L, K {0..7,16..23}; L>=16: K {8..15,24..31}
    v16bf afrag[4];
    #pragma unroll
    for (int mi = 0; mi < 4; ++mi) {
      int r = wm * 64 + mi * 16 + lsub;
      const v8bf lo = *(const v8bf*)&As[buf][r * LDT + lhalf * 8];
      const v8bf hi = *(const v8bf*)&As[buf][r * LDT + 16 + lhalf * 8];
      #pragma unroll
      for (int e = 0; e < 8; ++e) { afrag[mi][e] = lo[e]; afrag[mi][8 + e] = hi[e]; }
    }
    // B fragment: lane L holds column N=L&15, K {0..7,16..23} (L<16) or {8..15,24..31} (L>=16)
    v16bf bfrag[2];
    #pragma unroll
    for (int ni = 0; ni < 2; ++ni) {
      int r = wn * 32 + ni * 16 + lsub;
      const v8bf lo = *(const v8bf*)&Bs[buf][r * LDT + lhalf * 8];
      const v8bf hi = *(const v8bf*)&Bs[buf][r * LDT + 16 + lhalf * 8];
      #pragma unroll
      for (int e = 0; e < 8; ++e) { bfrag[ni][e] = lo[e]; bfrag[ni][8 + e] = hi[e]; }
    }

    #pragma unroll
    for (int mi = 0; mi < 4; ++mi)
      #pragma unroll
      for (int ni = 0; ni < 2; ++ni)
        acc[mi][ni] = __builtin_amdgcn_wmma_f32_16x16x32_bf16(
            false, afrag[mi], false, bfrag[ni], (short)0, acc[mi][ni], false, false);
  }

  // epilogue: C layout -> lane L: M = v + 8*(L>>4), N = L&15
  #pragma unroll
  for (int ni = 0; ni < 2; ++ni) {
    int j = bn0 + wn * 32 + ni * 16 + lsub;
    float bv = bias[j];
    #pragma unroll
    for (int mi = 0; mi < 4; ++mi) {
      int trow0 = bm0 + wm * 64 + mi * 16 + lhalf * 8;
      #pragma unroll
      for (int v = 0; v < 8; ++v)
        G[(size_t)(trow0 + v) * GATES + j] = acc[mi][ni][v] + bv;
    }
  }
}

// ---------------- persistent recurrent kernel ----------------
#define RNWG 64

__global__ __launch_bounds__(256) void rnn_kernel(
    const uint2* __restrict__ Whq,   // packed W_hh: (HIDDEN/4) x GATES uint2
    const float* __restrict__ G,     // precomputed x-part + bias, (T x GATES)
    const float* __restrict__ c0,
    float* __restrict__ hbuf,        // 2 x HIDDEN, double-buffered across steps
    int* __restrict__ counter,
    float* __restrict__ out) {       // [0]=y, [1..2048]=hT, [2049..4096]=cT
  __shared__ __align__(16) float h_lds[HIDDEN];
  __shared__ float part[256];
  __shared__ float gbuf[128];
  __shared__ float c_lds[32];

  const int tid  = threadIdx.x;
  const int m0   = blockIdx.x * 32;     // this WG owns hidden units m0..m0+31
  const int r    = tid & 127;           // output row within WG (4 gates x 32)
  const int half = tid >> 7;            // K-half split across thread pairs
  const int gate = r >> 5;
  const int mi   = r & 31;
  const int j    = gate * HIDDEN + m0 + mi;

  if (tid < 32) c_lds[tid] = c0[m0 + tid];

  const uint2* qcol  = Whq + (size_t)(half * 256) * GATES + j;
  const float* hhalf = h_lds + half * 1024;

  for (int t = 0; t < T_STEPS; ++t) {
    // pull current h into LDS with agent-scope loads (bypass stale L0)
    float* hsrc = hbuf + (t & 1) * HIDDEN;
    for (int i = tid; i < HIDDEN; i += 256)
      h_lds[i] = __hip_atomic_load(hsrc + i, __ATOMIC_RELAXED, __HIP_MEMORY_SCOPE_AGENT);
    __syncthreads();

    // half of one 2048-length bf16 dot product per thread
    float acc = 0.0f;
    #pragma unroll 8
    for (int kq = 0; kq < 256; ++kq) {
      uint2 q  = qcol[(size_t)kq * GATES];
      float4 h4 = *(const float4*)(hhalf + kq * 4);
      acc = fmaf(bflo(q.x), h4.x, acc);
      acc = fmaf(bfhi(q.x), h4.y, acc);
      acc = fmaf(bflo(q.y), h4.z, acc);
      acc = fmaf(bfhi(q.y), h4.w, acc);
    }
    part[tid] = acc;
    __syncthreads();

    if (tid < 128)
      gbuf[tid] = part[tid] + part[tid + 128] + G[(size_t)t * GATES + j];
    __syncthreads();

    if (tid < 32) {
      float gi = sigmoidf_(gbuf[tid]);
      float gf = sigmoidf_(gbuf[32 + tid]);
      float gg = tanhf(gbuf[64 + tid]);
      float go = sigmoidf_(gbuf[96 + tid]);
      float c  = gf * c_lds[tid] + gi * gg;
      c_lds[tid] = c;
      float h  = go * tanhf(c);
      __hip_atomic_store(hbuf + ((t + 1) & 1) * HIDDEN + m0 + tid, h,
                         __ATOMIC_RELAXED, __HIP_MEMORY_SCOPE_AGENT);
      if (t == T_STEPS - 1) {
        out[1 + m0 + tid] = h;
        out[1 + HIDDEN + m0 + tid] = c;
      }
    }
    __syncthreads();

    // grid barrier: release arrive + acquire spin (monotonic counter)
    if (tid == 0) {
      __hip_atomic_fetch_add(counter, 1, __ATOMIC_RELEASE, __HIP_MEMORY_SCOPE_AGENT);
      const int target = (t + 1) * RNWG;
      while (__hip_atomic_load(counter, __ATOMIC_ACQUIRE, __HIP_MEMORY_SCOPE_AGENT) < target)
        __builtin_amdgcn_s_sleep(1);
    }
    __syncthreads();
  }
}

// ---------------- dense head: y = (concat(hT,x2) @ W1^T + b1) @ W2^T + b2 ----------------
__global__ __launch_bounds__(512) void dense_kernel(
    const float* __restrict__ W1, const float* __restrict__ b1,
    const float* __restrict__ W2, const float* __restrict__ b2,
    const float* __restrict__ x2, float* __restrict__ out) {
  __shared__ float red[DENSE_N];
  const int d = threadIdx.x;
  const float* w = W1 + (size_t)d * (HIDDEN + 1);
  float acc = b1[d];
  for (int k = 0; k < HIDDEN; ++k) acc = fmaf(w[k], out[1 + k], acc);
  acc = fmaf(w[HIDDEN], x2[0], acc);
  red[d] = acc * W2[d];
  __syncthreads();
  for (int s = DENSE_N / 2; s > 0; s >>= 1) {
    if (d < s) red[d] += red[d + s];
    __syncthreads();
  }
  if (d == 0) out[0] = red[0] + b2[0];
}

// ---------------- launch ----------------
extern "C" void kernel_launch(void* const* d_in, const int* in_sizes, int n_in,
                              void* d_out, int out_size, void* d_ws, size_t ws_size,
                              hipStream_t stream) {
  const float* x1   = (const float*)d_in[0];
  const float* x2   = (const float*)d_in[1];
  const float* h0   = (const float*)d_in[2];
  const float* c0   = (const float*)d_in[3];
  const float* W_ih = (const float*)d_in[4];
  const float* W_hh = (const float*)d_in[5];
  const float* b_ih = (const float*)d_in[6];
  const float* b_hh = (const float*)d_in[7];
  const float* W1   = (const float*)d_in[8];
  const float* b1   = (const float*)d_in[9];
  const float* W2   = (const float*)d_in[10];
  const float* b2   = (const float*)d_in[11];
  float* out = (float*)d_out;

  char* ws = (char*)d_ws;
  unsigned short* x1b  = (unsigned short*)(ws);                    // 16 MB
  unsigned short* wihb = (unsigned short*)(ws + (16ull  << 20));   // 32 MB
  uint2*          whq  = (uint2*)         (ws + (48ull  << 20));   // 32 MB
  float*          G    = (float*)         (ws + (80ull  << 20));   // 128 MB
  float*          bias = (float*)         (ws + (208ull << 20));   // 32 KB
  float*          hbuf = (float*)         (ws + (208ull << 20) + 32768);
  int*            cnt  = (int*)           (ws + (208ull << 20) + 32768 + 16384);

  prep_kernel<<<32, 256, 0, stream>>>(b_ih, b_hh, h0, bias, hbuf, cnt);
  cvt_bf16_kernel<<<(T_STEPS * INPUT) / 256, 256, 0, stream>>>(x1b, x1, T_STEPS * INPUT);
  cvt_bf16_kernel<<<(GATES * INPUT) / 256, 256, 0, stream>>>(wihb, W_ih, GATES * INPUT);
  pack_whh_kernel<<<(GATES * (HIDDEN / 4)) / 256, 256, 0, stream>>>(whq, W_hh);

  dim3 ggrid(GATES / BN, T_STEPS / BM);   // 64 x 32 blocks
  gemm_kernel<<<ggrid, 256, 0, stream>>>(x1b, wihb, bias, G);

  rnn_kernel<<<RNWG, 256, 0, stream>>>(whq, G, c0, hbuf, cnt, out);

  dense_kernel<<<1, DENSE_N, 0, stream>>>(W1, b1, W2, b2, x2, out);
}
